// GNN_446676598862
// MI455X (gfx1250) — matrix operand order repaired and checked by reading, not compiled, not run on hardware
//
#include <hip/hip_runtime.h>
#include <hip/hip_bf16.h>

// ---------------- CDNA5 WMMA types ----------------
typedef __attribute__((ext_vector_type(16))) _Float16 v16h;
typedef __attribute__((ext_vector_type(8)))  float    v8f;

union F16Frag { v16h v; uint4 q[2]; };

#define BATCH 64
#define NN    64
#define FF    128
#define EE    4
#define MM    128
#define ROWS  (BATCH * NN)   // 4096
#define BIGF  1.0e6f

__device__ __forceinline__ float selu_f(float x) {
    const float scale = 1.0507009873554805f;
    const float alpha = 1.6732632423543772f;
    return x > 0.f ? scale * x : scale * alpha * (__expf(x) - 1.f);
}

// CDNA5 async Global -> LDS copy (16B per lane), tracked by ASYNCcnt.
__device__ __forceinline__ void async_load_b128(unsigned lds_off, const void* gptr) {
    asm volatile("global_load_async_to_lds_b128 %0, %1, off"
                 :: "v"(lds_off), "v"((unsigned long long)(size_t)gptr)
                 : "memory");
}
__device__ __forceinline__ void wait_asynccnt0() {
    asm volatile("s_wait_asynccnt 0x0" ::: "memory");
}

// ---------------- tiny helpers ----------------
__global__ void to_half_kernel(const float* __restrict__ x, _Float16* __restrict__ y) {
    int i = blockIdx.x * blockDim.x + threadIdx.x;
    y[i] = (_Float16)x[i];
}

// src: [E][128][128] f32 (K-major, W[k][n]); dst: [E][128][128] f16 N-major (W^T[n][k])
__global__ void xpose_w_kernel(const float* __restrict__ src, _Float16* __restrict__ dst) {
    int e = blockIdx.x;
    for (int i = threadIdx.x; i < 128 * 128; i += blockDim.x) {
        int k = i >> 7, n = i & 127;
        dst[e * 16384 + n * 128 + k] = (_Float16)src[e * 16384 + k * 128 + n];
    }
}

// ---------------- WMMA edge-type MLP: out[r][e][m] = SELU(h[r]·W0[e])·W1[e] ----------------
// grid: (64 row-tile groups, 4 edge types), block: 128 (4 waves, 1 row tile each)
__global__ __launch_bounds__(128) void edge_mlp_kernel(
        const _Float16* __restrict__ h16,   // [4096][128] row-major f16
        const _Float16* __restrict__ WT0,   // [E][128][128] N-major f16
        const _Float16* __restrict__ WT1,   // [E][128][128] N-major f16
        float* __restrict__ out)            // [4096][E][128] f32
{
    const int tid  = threadIdx.x;
    const int wave = tid >> 5;
    const int lane = tid & 31;
    const int lr   = lane & 15;   // row (A) or column (B/C) within tile
    const int hi   = lane >> 4;   // half-wave select
    const int e    = blockIdx.y;
    const int rt   = blockIdx.x * 4 + wave;
    const int r0   = rt << 4;

    __shared__ __align__(16) _Float16 sW[128 * 128];    // 32 KB block-shared weight panel
    __shared__ __align__(16) _Float16 sA[4][16 * 128];  // 16 KB per-wave A / hidden tiles

    const _Float16* __restrict__ B0g = WT0 + e * 16384;
    const _Float16* __restrict__ B1g = WT1 + e * 16384;

    // warm L2 for the layer-2 weight panel while layer 1 runs
    __builtin_prefetch((const char*)B1g + tid * 256, 0, 1);
    __builtin_prefetch((const char*)B1g + tid * 256 + 128, 0, 1);

    const unsigned sW_base = (unsigned)(size_t)(void*)&sW[0];
    const unsigned sA_base = (unsigned)(size_t)(void*)&sA[wave][0];

    // ---- async stage: W0^T (block-wide, 32KB) + A tile (per wave, 4KB) ----
#pragma unroll
    for (int it = 0; it < 16; ++it) {
        const unsigned off = (unsigned)(it * 2048 + tid * 16);
        async_load_b128(sW_base + off, (const char*)B0g + off);
    }
#pragma unroll
    for (int it = 0; it < 8; ++it) {
        const int c   = it * 32 + lane;          // 16-byte chunk id, 0..255
        const int row = c >> 4, col = c & 15;
        async_load_b128(sA_base + (unsigned)(c * 16),
                        (const void*)(h16 + (size_t)(r0 + row) * 128 + col * 8));
    }
    wait_asynccnt0();
    __syncthreads();

    // ---- layer 1: acc = h_tile @ W0[e] (A from LDS, B from LDS) ----
    v8f acc[8] = {};
    const _Float16* ap = &sA[wave][0];
#pragma unroll
    for (int ks = 0; ks < 4; ++ks) {
        F16Frag a;
        const int kb = ks * 32 + hi * 8;       // A layout: K halves {kb..kb+7, kb+16..kb+23}
        a.q[0] = *(const uint4*)(ap + lr * 128 + kb);
        a.q[1] = *(const uint4*)(ap + lr * 128 + kb + 16);
#pragma unroll
        for (int ct = 0; ct < 8; ++ct) {
            F16Frag bf;
            const _Float16* bp = sW + (ct * 16 + lr) * 128 + ks * 32 + hi * 8;
            bf.q[0] = *(const uint4*)(bp);
            bf.q[1] = *(const uint4*)(bp + 16);
            acc[ct] = __builtin_amdgcn_wmma_f32_16x16x32_f16(
                false, a.v, false, bf.v, (short)0, acc[ct], false, false);
        }
    }

    // ---- SELU + restage hidden tile (overwrites per-wave A tile) ----
    _Float16* hidp = &sA[wave][0];
#pragma unroll
    for (int ct = 0; ct < 8; ++ct)
#pragma unroll
        for (int j = 0; j < 8; ++j) {
            const int row = j + hi * 8;        // C layout: vgpr j -> M = j (+8 for upper lanes)
            const int col = ct * 16 + lr;      // N = lane % 16
            hidp[row * 128 + col] = (_Float16)selu_f(acc[ct][j]);
        }
    __syncthreads();   // everyone done reading W0 panel

    // ---- async stage: W1^T over the same panel ----
#pragma unroll
    for (int it = 0; it < 16; ++it) {
        const unsigned off = (unsigned)(it * 2048 + tid * 16);
        async_load_b128(sW_base + off, (const char*)B1g + off);
    }
    wait_asynccnt0();
    __syncthreads();

    // ---- layer 2: acc2 = hidden @ W1[e] ----
    v8f acc2[8] = {};
#pragma unroll
    for (int ks = 0; ks < 4; ++ks) {
        F16Frag a;
        const int kb = ks * 32 + hi * 8;
        a.q[0] = *(const uint4*)(ap + lr * 128 + kb);
        a.q[1] = *(const uint4*)(ap + lr * 128 + kb + 16);
#pragma unroll
        for (int ct = 0; ct < 8; ++ct) {
            F16Frag bf;
            const _Float16* bp = sW + (ct * 16 + lr) * 128 + ks * 32 + hi * 8;
            bf.q[0] = *(const uint4*)(bp);
            bf.q[1] = *(const uint4*)(bp + 16);
            acc2[ct] = __builtin_amdgcn_wmma_f32_16x16x32_f16(
                false, a.v, false, bf.v, (short)0, acc2[ct], false, false);
        }
    }

    // ---- store to out[row][e][m] f32 ----
#pragma unroll
    for (int ct = 0; ct < 8; ++ct)
#pragma unroll
        for (int j = 0; j < 8; ++j) {
            const int row = r0 + j + hi * 8;
            const int col = ct * 16 + lr;
            out[((size_t)row * EE + e) * MM + col] = acc2[ct][j];
        }
}

// ---------------- attention softmax + message + GRU update ----------------
// grid: (N, B), block: 128 (thread = feature m)
__global__ __launch_bounds__(128) void attn_gru_kernel(
        const float* __restrict__ adj, const float* __restrict__ ed,
        const float* __restrict__ phi, const float* __restrict__ psi,
        const float* __restrict__ Wz,  const float* __restrict__ Wn,
        float* __restrict__ h)
{
    const int v = blockIdx.x, b = blockIdx.y;
    const int m = threadIdx.x;
    __shared__ float sAdj[NN];
    __shared__ float sEd[NN * EE];
    __shared__ float sMsg[MM];

    const float* adjrow = adj + ((size_t)b * NN + v) * NN;
    const float* edrow  = ed  + (((size_t)b * NN + v) * NN) * EE;
    if (m < NN) sAdj[m] = adjrow[m];
    sEd[m]       = edrow[m];
    sEd[m + 128] = edrow[m + 128];
    __syncthreads();

    // online softmax over neighbours u
    float mx = -3.0e38f, s = 0.f, acc = 0.f;
    for (int u = 0; u < NN; ++u) {
        if (sAdj[u] != 0.f) {
            const float* pr = psi + (((size_t)b * NN + u) * EE) * MM + m;
            const float* fr = phi + (((size_t)b * NN + u) * EE) * MM + m;
            const float e0 = sEd[u * 4 + 0], e1 = sEd[u * 4 + 1];
            const float e2 = sEd[u * 4 + 2], e3 = sEd[u * 4 + 3];
            const float pv = e0 * pr[0] + e1 * pr[128] + e2 * pr[256] + e3 * pr[384];
            const float fv = e0 * fr[0] + e1 * fr[128] + e2 * fr[256] + e3 * fr[384];
            const float nm   = fmaxf(mx, pv);
            const float corr = __expf(mx - nm);
            const float p    = __expf(pv - nm);
            s   = s * corr + p;
            acc = acc * corr + p * fv;
            mx  = nm;
        }
    }
    sMsg[m] = (s > 0.f) ? acc / s : 0.f;

    float deg = 0.f;
    for (int u = 0; u < NN; ++u) deg += sAdj[u];
    __syncthreads();

    // GRUCell(bias=False, h_prev=0): h' = (1 - sigmoid(msg·Wz)) * tanh(msg·Wn)
    float z = 0.f, n = 0.f;
    for (int k = 0; k < MM; ++k) {
        const float mk = sMsg[k];
        z += mk * Wz[k * FF + m];
        n += mk * Wn[k * FF + m];
    }
    const float zs = 1.f / (1.f + __expf(-z));
    const float nh = (1.f - zs) * tanhf(n);
    if (deg > 0.f) h[((size_t)b * NN + v) * FF + m] = nh;
}

// ---------------- graph-gather readout ----------------
// grid: B, block: 128 (thread = feature)
__global__ __launch_bounds__(128) void readout_kernel(
        const float* __restrict__ h, const float* __restrict__ nd,
        const float* __restrict__ adj,
        const float* __restrict__ gaW0, const float* __restrict__ gaW1,
        const float* __restrict__ geW0, const float* __restrict__ geW1,
        float* __restrict__ g)
{
    const int b = blockIdx.x, f = threadIdx.x;
    __shared__ float sCat[2 * FF];
    __shared__ float sA1[128];
    __shared__ float sE1[128];
    __shared__ float sEMask[NN];

    if (f < NN) {
        float d = 0.f;
        const float* ar = adj + ((size_t)b * NN + f) * NN;
        for (int u = 0; u < NN; ++u) d += ar[u];
        sEMask[f] = (d > 0.f) ? 0.f : BIGF;
    }
    float gacc = 0.f;
    __syncthreads();

    for (int v = 0; v < NN; ++v) {
        sCat[f]       = h [((size_t)b * NN + v) * FF + f];
        sCat[FF + f]  = nd[((size_t)b * NN + v) * FF + f];
        __syncthreads();
        float a1 = 0.f;
        for (int k = 0; k < 2 * FF; ++k) a1 += sCat[k] * gaW0[k * 128 + f];
        float e1 = 0.f;
        for (int k = 0; k < FF; ++k)     e1 += sCat[k] * geW0[k * 128 + f];
        sA1[f] = selu_f(a1);
        sE1[f] = selu_f(e1);
        __syncthreads();
        float att = 0.f, em = 0.f;
        for (int k = 0; k < 128; ++k) {
            att += sA1[k] * gaW1[k * 128 + f];
            em  += sE1[k] * geW1[k * 128 + f];
        }
        att = 1.f / (1.f + __expf(-(att - sEMask[v])));
        gacc += att * em;
        __syncthreads();
    }
    g[(size_t)b * 128 + f] = gacc;
}

// ---------------- final MLP ----------------
// grid: B, block: 256
__global__ __launch_bounds__(256) void final_mlp_kernel(
        const float* __restrict__ g1, const float* __restrict__ g2,
        const float* __restrict__ W0, const float* __restrict__ W1,
        float* __restrict__ out)
{
    const int b = blockIdx.x, t = threadIdx.x;
    __shared__ float sCat[256];
    __shared__ float sHid[256];
    sCat[t] = (t < 128) ? g1[(size_t)b * 128 + t] : g2[(size_t)b * 128 + (t - 128)];
    __syncthreads();
    float a = 0.f;
    for (int k = 0; k < 256; ++k) a += sCat[k] * W0[k * 256 + t];
    sHid[t] = selu_f(a);
    __syncthreads();
    if (t < 128) {
        float o = 0.f;
        for (int k = 0; k < 256; ++k) o += sHid[k] * W1[k * 128 + t];
        out[(size_t)b * 128 + t] = o;
    }
}

// ---------------- orchestration ----------------
extern "C" void kernel_launch(void* const* d_in, const int* in_sizes, int n_in,
                              void* d_out, int out_size, void* d_ws, size_t ws_size,
                              hipStream_t stream) {
    (void)in_sizes; (void)n_in; (void)out_size; (void)ws_size;

    const float* adj[2] = {(const float*)d_in[0], (const float*)d_in[3]};
    const float* nd[2]  = {(const float*)d_in[1], (const float*)d_in[4]};
    const float* ed[2]  = {(const float*)d_in[2], (const float*)d_in[5]};

    char* ws = (char*)d_ws;
    size_t off = 0;
    auto alloc = [&](size_t bytes) -> void* {
        void* p = ws + off;
        off = (off + bytes + 255) & ~(size_t)255;
        return p;
    };
    float*    h    = (float*)   alloc((size_t)ROWS * FF * 4);         // 2 MB
    _Float16* h16  = (_Float16*)alloc((size_t)ROWS * FF * 2);         // 1 MB
    float*    phi  = (float*)   alloc((size_t)ROWS * EE * MM * 4);    // 8 MB
    float*    psi  = (float*)   alloc((size_t)ROWS * EE * MM * 4);    // 8 MB
    _Float16* mWT0 = (_Float16*)alloc((size_t)EE * 128 * 128 * 2);
    _Float16* mWT1 = (_Float16*)alloc((size_t)EE * 128 * 128 * 2);
    _Float16* aWT0 = (_Float16*)alloc((size_t)EE * 128 * 128 * 2);
    _Float16* aWT1 = (_Float16*)alloc((size_t)EE * 128 * 128 * 2);
    float*    g    = (float*)   alloc((size_t)2 * BATCH * 128 * 4);

    for (int t = 0; t < 2; ++t) {
        const int wb = 6 + 10 * t;
        const float* mW0  = (const float*)d_in[wb + 0];
        const float* mW1  = (const float*)d_in[wb + 1];
        const float* aW0  = (const float*)d_in[wb + 2];
        const float* aW1  = (const float*)d_in[wb + 3];
        const float* Wz   = (const float*)d_in[wb + 4];
        const float* Wn   = (const float*)d_in[wb + 5];
        const float* gaW0 = (const float*)d_in[wb + 6];
        const float* gaW1 = (const float*)d_in[wb + 7];
        const float* geW0 = (const float*)d_in[wb + 8];
        const float* geW1 = (const float*)d_in[wb + 9];

        hipMemcpyAsync(h, nd[t], (size_t)ROWS * FF * 4, hipMemcpyDeviceToDevice, stream);

        xpose_w_kernel<<<dim3(EE), 256, 0, stream>>>(mW0, mWT0);
        xpose_w_kernel<<<dim3(EE), 256, 0, stream>>>(mW1, mWT1);
        xpose_w_kernel<<<dim3(EE), 256, 0, stream>>>(aW0, aWT0);
        xpose_w_kernel<<<dim3(EE), 256, 0, stream>>>(aW1, aWT1);

        for (int pass = 0; pass < 2; ++pass) {
            to_half_kernel<<<dim3(ROWS * FF / 256), 256, 0, stream>>>(h, h16);
            edge_mlp_kernel<<<dim3(64, EE), 128, 0, stream>>>(h16, mWT0, mWT1, phi);
            edge_mlp_kernel<<<dim3(64, EE), 128, 0, stream>>>(h16, aWT0, aWT1, psi);
            attn_gru_kernel<<<dim3(NN, BATCH), 128, 0, stream>>>(
                adj[t], ed[t], phi, psi, Wz, Wn, h);
        }
        readout_kernel<<<dim3(BATCH), 128, 0, stream>>>(
            h, nd[t], adj[t], gaW0, gaW1, geW0, geW1, g + (size_t)t * BATCH * 128);
    }

    final_mlp_kernel<<<dim3(BATCH), 256, 0, stream>>>(
        g, g + (size_t)BATCH * 128,
        (const float*)d_in[26], (const float*)d_in[27], (float*)d_out);
}
